// relationalGraphStack_781684048166
// MI455X (gfx1250) — matrix operand order repaired and checked by reading, not compiled, not run on hardware
//
#include <hip/hip_runtime.h>
#include <hip/hip_bf16.h>

typedef _Float16 half_t;
typedef __attribute__((ext_vector_type(16))) _Float16 v16h;
typedef __attribute__((ext_vector_type(8)))  float    v8f;
typedef __attribute__((ext_vector_type(4)))  float    v4f;

#define GN 50000
#define GD 128
#define GR 8
#define GE 800000

// ---------------------------------------------------------------- utilities
__global__ void k_zero(float* __restrict__ p, long long n) {
  long long i = (long long)blockIdx.x * blockDim.x + threadIdx.x;
  if (i < n) p[i] = 0.0f;
}

__global__ void k_deg(const int* __restrict__ node_out, const int* __restrict__ relation,
                      const float* __restrict__ w, float* __restrict__ deg, int E) {
  int e = blockIdx.x * blockDim.x + threadIdx.x;
  if (e < E) atomicAdd(&deg[node_out[e] * GR + relation[e]], w[e]);
}

// Pack f32 weight [K,128] row-major into f16 WMMA B fragments.
// Fragment (kt,nt): 32 lanes x 16 halves contiguous.
// B layout (16-bit 32x16): lane L -> column n = L%16; element j -> K = (j/8)*16 + (L/16)*8 + (j%8).
__global__ void k_pack_b(const float* __restrict__ W, half_t* __restrict__ P, int ktiles) {
  long long idx = (long long)blockIdx.x * blockDim.x + threadIdx.x;
  long long total = (long long)ktiles * 4096;
  if (idx >= total) return;
  int j    = (int)(idx & 15);
  int lane = (int)((idx >> 4) & 31);
  long long frag = idx >> 9;
  int nt = (int)(frag & 7);
  int kt = (int)(frag >> 3);
  int k = kt * 32 + (j >> 3) * 16 + (lane >> 4) * 8 + (j & 7);
  int n = nt * 16 + (lane & 15);
  P[idx] = (half_t)W[(size_t)k * GD + n];
}

// Build 16-bit A fragment (16x32) from a contiguous f32 row pointer.
// lane L: row M = L%16 ; element j -> K = (j/8)*16 + (L/16)*8 + (j%8)
__device__ inline v16h load_a_frag(const float* __restrict__ rowp, int g) {
  const float* p = rowp + g * 8;
  v4f q0 = *(const v4f*)(p);
  v4f q1 = *(const v4f*)(p + 4);
  v4f q2 = *(const v4f*)(p + 16);
  v4f q3 = *(const v4f*)(p + 20);
  v16h a;
#pragma unroll
  for (int j = 0; j < 4; ++j) {
    a[j]      = (_Float16)q0[j];
    a[j + 4]  = (_Float16)q1[j];
    a[j + 8]  = (_Float16)q2[j];
    a[j + 12] = (_Float16)q3[j];
  }
  return a;
}

// ------------------------------------------------- fused edge transform + scatter
// Per wave: 16 edges x 128 cols.  msg = ew * (src[node_in] + ef@We + be),
// atomic scatter-add into uout[node_out*sN + relation*sR].
__global__ void k_edge_msgs(const float* __restrict__ src, const float* __restrict__ ef,
                            const int* __restrict__ node_in, const int* __restrict__ node_out,
                            const int* __restrict__ relation, const float* __restrict__ ewt,
                            const float* __restrict__ deg, const half_t* __restrict__ Bw,
                            const float* __restrict__ bedge, float* __restrict__ uout,
                            int sN, int sR, int ntiles) {
  int lane = threadIdx.x & 31;
  int wave = threadIdx.x >> 5;
  int tile = blockIdx.x * (blockDim.x >> 5) + wave;
  if (tile >= ntiles) return;
  int g = lane >> 4;
  int m = lane & 15;
  int e = tile * 16 + m;

  int ni = node_in[e];
  int no = node_out[e];
  int rl = relation[e];
  float ev = ewt[e] / deg[no * GR + rl];
  int oi = no * sN + rl * sR;

  // broadcast per-row (per-edge) scalars to match D-matrix row layout M = r + 8*g
  int nia[8], oia[8];
  float eva[8];
#pragma unroll
  for (int r2 = 0; r2 < 8; ++r2) {
    int mm = r2 + 8 * g;
    nia[r2] = __shfl(ni, mm, 32);
    oia[r2] = __shfl(oi, mm, 32);
    eva[r2] = __shfl(ev, mm, 32);
  }

  v16h a = load_a_frag(ef + (size_t)e * 32, g);  // K = ED = 32 in one WMMA
  v8f zero = {};
#pragma unroll
  for (int nt = 0; nt < 8; ++nt) {
    v16h b = *(const v16h*)(Bw + ((size_t)nt * 32 + lane) * 16);
    v8f acc = __builtin_amdgcn_wmma_f32_16x16x32_f16(false, a, false, b, (short)0, zero,
                                                     false, false);
    int n = nt * 16 + m;
    float be = bedge[n];
#pragma unroll
    for (int r2 = 0; r2 < 8; ++r2) {
      float msg = eva[r2] * (src[(size_t)nia[r2] * GD + n] + acc[r2] + be);
      atomicAdd(uout + (size_t)oia[r2] * GD + n, msg);
    }
  }
}

// ------------------------------------------------------------- combine GEMM
// out[row,:] = concat(A0[row,:k0*32], A1[row%rowmod,:]) @ B + bias0 + bias1
// One 16x128 tile per wave, K loop in steps of 32 with f16 WMMA, f32 accumulate.
__global__ void k_gemm(const float* __restrict__ A0, int k0tiles, int lda0,
                       const float* __restrict__ A1, int lda1, int rowmod, int ktiles,
                       const half_t* __restrict__ B, const float* __restrict__ bias0,
                       const float* __restrict__ bias1, float* __restrict__ out, int M) {
  int lane = threadIdx.x & 31;
  int wave = threadIdx.x >> 5;
  int tile = blockIdx.x * (blockDim.x >> 5) + wave;
  int row0 = tile * 16;
  if (row0 >= M) return;
  int g = lane >> 4;
  int m = lane & 15;
  int row = row0 + m;

  v8f zero = {};
  v8f acc[8];
#pragma unroll
  for (int i = 0; i < 8; ++i) acc[i] = zero;

  for (int kt = 0; kt < ktiles; ++kt) {
    const float* ap;
    if (kt < k0tiles)
      ap = A0 + (size_t)row * lda0 + kt * 32;
    else
      ap = A1 + (size_t)(row % rowmod) * lda1 + (kt - k0tiles) * 32;
    __builtin_prefetch(ap + 32, 0, 1);
    v16h a = load_a_frag(ap, g);
    const half_t* bbase = B + ((size_t)kt * 256 + lane) * 16;
#pragma unroll
    for (int nt = 0; nt < 8; ++nt) {
      v16h b = *(const v16h*)(bbase + (size_t)nt * 512);
      acc[nt] = __builtin_amdgcn_wmma_f32_16x16x32_f16(false, a, false, b, (short)0,
                                                       acc[nt], false, false);
    }
  }
#pragma unroll
  for (int nt = 0; nt < 8; ++nt) {
    int n = nt * 16 + m;
    float bs = bias0[n] + bias1[n];
#pragma unroll
    for (int r2 = 0; r2 < 8; ++r2) {
      int orow = row0 + r2 + 8 * g;
      out[(size_t)orow * GD + n] = acc[nt][r2] + bs;
    }
  }
}

// ----------------------------------------------------------------- batch norm
__global__ void k_bn_stats(const float* __restrict__ X, long long M, int rpb,
                           float* __restrict__ sums) {
  int col = threadIdx.x;  // blockDim = 128
  long long r0 = (long long)blockIdx.x * rpb;
  long long r1 = r0 + rpb;
  if (r1 > M) r1 = M;
  float s = 0.0f, s2 = 0.0f;
  for (long long r = r0; r < r1; ++r) {
    float v = X[r * GD + col];
    s += v;
    s2 += v * v;
  }
  atomicAdd(&sums[col], s);
  atomicAdd(&sums[GD + col], s2);
}

__global__ void k_bn_finalize(const float* __restrict__ sums, float Minv,
                              const float* __restrict__ gamma, const float* __restrict__ beta,
                              float* __restrict__ ss) {
  int col = threadIdx.x;  // 128
  float mean = sums[col] * Minv;
  float var = sums[GD + col] * Minv - mean * mean;
  float inv = rsqrtf(var + 1e-5f);
  float sc = gamma[col] * inv;
  ss[col] = sc;
  ss[GD + col] = beta[col] - mean * sc;
}

__global__ void k_bn_apply(float* __restrict__ X, long long n, const float* __restrict__ ss) {
  long long i = (long long)blockIdx.x * blockDim.x + threadIdx.x;
  if (i >= n) return;
  int col = (int)(i & (GD - 1));
  float v = X[i] * ss[col] + ss[GD + col];
  X[i] = v > 0.0f ? v : 0.0f;
}

// ----------------------------------------------------------------------------
extern "C" void kernel_launch(void* const* d_in, const int* in_sizes, int n_in,
                              void* d_out, int out_size, void* d_ws, size_t ws_size,
                              hipStream_t stream) {
  (void)in_sizes; (void)n_in; (void)out_size; (void)ws_size;
  const float* x            = (const float*)d_in[0];
  const int*   node_in      = (const int*)d_in[1];
  const int*   node_out     = (const int*)d_in[2];
  const int*   relation     = (const int*)d_in[3];
  const float* edge_weight  = (const float*)d_in[4];
  const float* edge_feature = (const float*)d_in[5];
  const float* l1_Wlin  = (const float*)d_in[6];
  const float* l1_blin  = (const float*)d_in[7];
  const float* l1_Wself = (const float*)d_in[8];
  const float* l1_bself = (const float*)d_in[9];
  const float* l1_Wedge = (const float*)d_in[10];
  const float* l1_bedge = (const float*)d_in[11];
  const float* l1_g     = (const float*)d_in[12];
  const float* l1_b     = (const float*)d_in[13];
  const float* l2_Wlin  = (const float*)d_in[14];
  const float* l2_blin  = (const float*)d_in[15];
  const float* l2_Wself = (const float*)d_in[16];
  const float* l2_bself = (const float*)d_in[17];
  const float* l2_Wedge = (const float*)d_in[18];
  const float* l2_bedge = (const float*)d_in[19];
  const float* l2_g     = (const float*)d_in[20];
  const float* l2_b     = (const float*)d_in[21];
  float* out = (float*)d_out;

  const long long NR  = (long long)GN * GR;          // 400000
  const long long NRD = NR * GD;                     // 51.2M
  const long long ND  = (long long)GN * GD;          // 6.4M

  char* ws = (char*)d_ws;
  size_t off = 0;
  auto alloc = [&](size_t bytes) -> void* {
    void* p = ws + off;
    off += (bytes + 255) & ~(size_t)255;
    return p;
  };
  float*  u    = (float*)alloc((size_t)NRD * 4);     // shared by layer1/layer2 aggregation
  float*  deg  = (float*)alloc((size_t)NR * 4);
  float*  h    = (float*)alloc((size_t)ND * 4);
  float*  sums = (float*)alloc(256 * 4);
  float*  ss   = (float*)alloc(256 * 4);
  half_t* pW1  = (half_t*)alloc((size_t)36 * 4096 * 2);  // [Wlin(1024); Wself(128)] K=1152
  half_t* pW2  = (half_t*)alloc((size_t)8 * 4096 * 2);   // [Wlin(128); Wself(128)]  K=256
  half_t* pE1  = (half_t*)alloc((size_t)4096 * 2);
  half_t* pE2  = (half_t*)alloc((size_t)4096 * 2);

  // --- init + weight packing ------------------------------------------------
  k_zero<<<(unsigned)((NRD + 255) / 256), 256, 0, stream>>>(u, NRD);
  k_zero<<<(unsigned)((NR + 255) / 256), 256, 0, stream>>>(deg, NR);
  k_zero<<<1, 256, 0, stream>>>(sums, 256);
  k_pack_b<<<32 * 16, 256, 0, stream>>>(l1_Wlin, pW1, 32);
  k_pack_b<<<4 * 16, 256, 0, stream>>>(l1_Wself, pW1 + (size_t)32 * 4096, 4);
  k_pack_b<<<4 * 16, 256, 0, stream>>>(l2_Wlin, pW2, 4);
  k_pack_b<<<4 * 16, 256, 0, stream>>>(l2_Wself, pW2 + (size_t)4 * 4096, 4);
  k_pack_b<<<16, 256, 0, stream>>>(l1_Wedge, pE1, 1);
  k_pack_b<<<16, 256, 0, stream>>>(l2_Wedge, pE2, 1);

  // --- degree normalization -------------------------------------------------
  k_deg<<<GE / 256, 256, 0, stream>>>(node_out, relation, edge_weight, deg, GE);

  // --- layer 1 --------------------------------------------------------------
  // fused edge GEMM + gather + scatter into u[(node_out*R+relation), :]
  k_edge_msgs<<<GE / 16 / 8, 256, 0, stream>>>(x, edge_feature, node_in, node_out, relation,
                                               edge_weight, deg, pE1, l1_bedge, u, GR, 1,
                                               GE / 16);
  // h = u1.reshape(N,1024) @ Wlin + x @ Wself + biases   (K = 1152)
  k_gemm<<<(GN / 16 + 7) / 8, 256, 0, stream>>>(u, 32, 1024, x, GD, GN, 36, pW1, l1_blin,
                                                l1_bself, h, GN);
  k_bn_stats<<<(GN + 255) / 256, 128, 0, stream>>>(h, GN, 256, sums);
  k_bn_finalize<<<1, 128, 0, stream>>>(sums, 1.0f / GN, l1_g, l1_b, ss);
  k_bn_apply<<<(unsigned)((ND + 255) / 256), 256, 0, stream>>>(h, ND, ss);

  // --- layer 2 --------------------------------------------------------------
  k_zero<<<(unsigned)((NRD + 255) / 256), 256, 0, stream>>>(u, NRD);
  k_zero<<<1, 256, 0, stream>>>(sums, 256);
  // scatter directly at the permuted index relation*N + node_out (== transpose(1,0,2))
  k_edge_msgs<<<GE / 16 / 8, 256, 0, stream>>>(h, edge_feature, node_in, node_out, relation,
                                               edge_weight, deg, pE2, l2_bedge, u, 1, GN,
                                               GE / 16);
  // o = u2 @ Wlin + h[row % N] @ Wself + biases   (K = 256), written straight to d_out
  k_gemm<<<(int)(NR / 16 / 8), 256, 0, stream>>>(u, 4, GD, h, GD, GN, 8, pW2, l2_blin,
                                                 l2_bself, out, (int)NR);
  k_bn_stats<<<(unsigned)((NR + 255) / 256), 128, 0, stream>>>(out, NR, 256, sums);
  k_bn_finalize<<<1, 128, 0, stream>>>(sums, 1.0f / (float)NR, l2_g, l2_b, ss);
  k_bn_apply<<<(unsigned)((NRD + 255) / 256), 256, 0, stream>>>(out, NRD, ss);
}
